// FusionAttention_335007449971
// MI455X (gfx1250) — matrix operand order repaired and checked by reading, not compiled, not run on hardware
//
#include <hip/hip_runtime.h>
#include <hip/hip_bf16.h>

typedef __bf16 bf16;
typedef __attribute__((ext_vector_type(16))) __bf16 v16bf;
typedef __attribute__((ext_vector_type(8)))  __bf16 bf16x8;
typedef __attribute__((ext_vector_type(8)))  float  v8f;

#define DIM 256
#define SSM 256
#define NH  16
#define HD  16
#define WSZ 8
#define BB  2
#define HH  256
#define WW  256

#if __has_builtin(__builtin_amdgcn_permlane16)
#define FA_PERMLANE 1
#endif
#if __has_builtin(__builtin_amdgcn_global_load_async_to_lds_b128) && \
    __has_builtin(__builtin_amdgcn_s_wait_asynccnt)
#define FA_ASYNC 1
#endif

#ifdef FA_ASYNC
typedef int v4i __attribute__((vector_size(16)));
typedef v4i __attribute__((address_space(1)))* as1_v4i_p;
typedef v4i __attribute__((address_space(3)))* as3_v4i_p;
#endif

static __device__ __forceinline__ v8f fzero8() {
  v8f z;
#pragma unroll
  for (int i = 0; i < 8; ++i) z[i] = 0.0f;
  return z;
}
static __device__ __forceinline__ bf16x8 bzero8() {
  bf16x8 z;
#pragma unroll
  for (int i = 0; i < 8; ++i) z[i] = (bf16)0.0f;
  return z;
}
static __device__ __forceinline__ v16bf zero16() {
  v16bf z;
#pragma unroll
  for (int i = 0; i < 16; ++i) z[i] = (bf16)0.0f;
  return z;
}
static __device__ __forceinline__ v16bf cat16(bf16x8 lo, bf16x8 hi) {
  union { v16bf v; bf16x8 h[2]; } u;
  u.h[0] = lo; u.h[1] = hi;
  return u.v;
}
static __device__ __forceinline__ v16bf ld_a16(const bf16* lo, const bf16* hi) {
  return cat16(*(const bf16x8*)lo, *(const bf16x8*)hi);
}
static __device__ __forceinline__ v8f wmma_bf(v16bf a, v16bf b, v8f c) {
  return __builtin_amdgcn_wmma_f32_16x16x32_bf16(false, a, false, b, (short)0, c, false, false);
}

// ---- wave32 16-lane butterfly reductions (VALU permlane, DS fallback) ----
#ifdef FA_PERMLANE
static __device__ __forceinline__ float pl16f(float v, unsigned s0, unsigned s1) {
  unsigned u = __float_as_uint(v);
  return __uint_as_float(__builtin_amdgcn_permlane16(u, u, s0, s1, false, false));
}
#endif
static __device__ __forceinline__ float red16_max(float v) {
#ifdef FA_PERMLANE
  v = fmaxf(v, pl16f(v, 0x67452301u, 0xEFCDAB89u)); // xor 1
  v = fmaxf(v, pl16f(v, 0x45670123u, 0xCDEF89ABu)); // xor 2
  v = fmaxf(v, pl16f(v, 0x01234567u, 0x89ABCDEFu)); // xor 4
  v = fmaxf(v, pl16f(v, 0xFEDCBA98u, 0x76543210u)); // xor 8
#else
  for (int off = 1; off < 16; off <<= 1) v = fmaxf(v, __shfl_xor(v, off, 32));
#endif
  return v;
}
static __device__ __forceinline__ float red16_sum(float v) {
#ifdef FA_PERMLANE
  v += pl16f(v, 0x67452301u, 0xEFCDAB89u);
  v += pl16f(v, 0x45670123u, 0xCDEF89ABu);
  v += pl16f(v, 0x01234567u, 0x89ABCDEFu);
  v += pl16f(v, 0xFEDCBA98u, 0x76543210u);
#else
  for (int off = 1; off < 16; off <<= 1) v += __shfl_xor(v, off, 32);
#endif
  return v;
}

// ---------------- workspace layout (bytes) ----------------
#define NELM ((size_t)BB * DIM * HH * WW)           // 33,554,432
#define OFF_ATTN  ((size_t)0)
#define OFF_LOCAL (OFF_ATTN  + NELM * 2)
#define OFF_SUM   (OFF_LOCAL + NELM * 2)
#define OFF_DW    (OFF_SUM   + NELM * 2)
#define OFF_WQKV  (OFF_DW    + NELM * 2)
#define OFF_WL1   (OFF_WQKV  + (size_t)768 * 256 * 2)
#define OFF_WL2   (OFF_WL1   + (size_t)9 * 256 * 256 * 2)
#define OFF_WPW   (OFF_WL2   + (size_t)256 * 256 * 2)
#define OFF_WDW   (OFF_WPW   + (size_t)256 * 256 * 2)
#define OFF_CLOC  (OFF_WDW   + (size_t)256 * 64 * 4)
#define OFF_CDW   (OFF_CLOC  + (size_t)256 * 4)

static __device__ __forceinline__ size_t bpack_idx(int nTile, int kTile, int n, int k, int kTilesPerN) {
  int lane = (((k >> 3) & 1) << 4) | (n & 15);
  int pos = (k & 7) | (((k >> 4) & 1) << 3);
  return ((size_t)(nTile * kTilesPerN + kTile) * 32 + lane) * 16 + pos;
}

// ================= K0: pack weights (BN folded) =================
__global__ void fa_pack(const float* wqkv, const float* wl1,
                        const float* g1, const float* b1, const float* m1, const float* v1,
                        const float* wl2,
                        const float* g2, const float* b2, const float* m2, const float* v2,
                        const float* wdw,
                        const float* gp, const float* bp, const float* mp, const float* vp,
                        const float* wpw,
                        bf16* pqkv, bf16* pl1, bf16* pl2, bf16* ppw,
                        float* pdw, float* cloc, float* cdw) {
  int i = blockIdx.x * 256 + threadIdx.x;
  const float EPS = 1e-5f;
  if (i < 768 * 256) {
    int n = i >> 8, k = i & 255;
    pqkv[bpack_idx(n >> 4, k >> 5, n, k, 8)] = (bf16)wqkv[i];
    return;
  }
  i -= 768 * 256;
  if (i < 256 * 256 * 9) {
    int co = i / (256 * 9);
    int rem = i - co * (256 * 9);
    int ci = rem / 9, tap = rem - ci * 9;
    float inv = g1[co] * rsqrtf(v1[co] + EPS);
    size_t dst = ((size_t)(tap * 16 + (co >> 4)) * 8 + (ci >> 5)) * 32;
    dst = (dst + ((((ci >> 3) & 1) << 4) | (co & 15))) * 16 + ((ci & 7) | (((ci >> 4) & 1) << 3));
    pl1[dst] = (bf16)(wl1[(co * 256 + ci) * 9 + tap] * inv);
    return;
  }
  i -= 256 * 256 * 9;
  if (i < 256 * 256) {
    int co = i >> 8, ci = i & 255;
    float inv = g2[co] * rsqrtf(v2[co] + EPS);
    pl2[bpack_idx(co >> 4, ci >> 5, co, ci, 8)] = (bf16)(wl2[i] * inv);
    return;
  }
  i -= 256 * 256;
  if (i < 256 * 256) {
    int n = i >> 8, k = i & 255;
    ppw[bpack_idx(n >> 4, k >> 5, n, k, 8)] = (bf16)wpw[i];
    return;
  }
  i -= 256 * 256;
  if (i < 256 * 64) {
    int c = i >> 6;
    float inv = gp[c] * rsqrtf(vp[c] + EPS);
    pdw[i] = wdw[i] * inv;
    return;
  }
  i -= 256 * 64;
  if (i < 256) {
    float i1 = g1[i] * rsqrtf(v1[i] + EPS);
    float i2 = g2[i] * rsqrtf(v2[i] + EPS);
    float ip = gp[i] * rsqrtf(vp[i] + EPS);
    cloc[i] = (b1[i] - m1[i] * i1) + (b2[i] - m2[i] * i2);
    cdw[i] = bp[i] - mp[i] * ip;
  }
}

// ================= K1: fused QKV + windowed attention =================
__global__ void __launch_bounds__(64) fa_attn(const float* __restrict__ x,
                                              const bf16* __restrict__ pqkv,
                                              const float* __restrict__ rpb,
                                              bf16* __restrict__ attn) {
  __shared__ __attribute__((aligned(32))) bf16 Xs[64 * 256];
  __shared__ __attribute__((aligned(32))) bf16 Qs[2][64 * 16];
  __shared__ __attribute__((aligned(32))) bf16 Ks[2][64 * 16];
  __shared__ __attribute__((aligned(32))) bf16 Vt[2][16 * 64];
  __shared__ __attribute__((aligned(32))) bf16 At[2][64 * 64];

  const int tid = threadIdx.x;
  const int wv = tid >> 5, lane = tid & 31, lh = lane & 15, hiq = (lane >> 4) & 1;
  const int win = blockIdx.x;
  const int wx = win & 31, wy = (win >> 5) & 31, b = win >> 10;
  const int h0 = wy * 8, w0 = wx * 8;

  if (tid == 0) __builtin_prefetch(rpb, 0, 0);

  for (int r = tid; r < 2048; r += 64) {
    int c = r >> 3, py = r & 7;
    const float4* src = (const float4*)(x + (((size_t)b * 256 + c) * 256 + (h0 + py)) * 256 + w0);
    float4 f0 = src[0], f1 = src[1];
    bf16* dst = &Xs[(py * 8) * 256 + c];
    dst[0 * 256] = (bf16)f0.x; dst[1 * 256] = (bf16)f0.y;
    dst[2 * 256] = (bf16)f0.z; dst[3 * 256] = (bf16)f0.w;
    dst[4 * 256] = (bf16)f1.x; dst[5 * 256] = (bf16)f1.y;
    dst[6 * 256] = (bf16)f1.z; dst[7 * 256] = (bf16)f1.w;
  }
  __syncthreads();

  for (int hh = 0; hh < 8; ++hh) {
    const int h = wv * 8 + hh;
    // ---- QKV projections ----
#pragma unroll
    for (int M = 0; M < 3; ++M) {
      const int nTile = M * 16 + h;
#pragma unroll
      for (int mt = 0; mt < 4; ++mt) {
        v8f acc = fzero8();
        const int row = mt * 16 + lh;
#pragma unroll
        for (int kt = 0; kt < 8; ++kt) {
          const int base = kt * 32 + hiq * 8;
          v16bf a = ld_a16(&Xs[row * 256 + base], &Xs[row * 256 + base + 16]);
          v16bf bm = *(const v16bf*)(pqkv + ((size_t)(nTile * 8 + kt) * 32 + lane) * 16);
          acc = wmma_bf(a, bm, acc);
        }
#pragma unroll
        for (int r = 0; r < 8; ++r) {
          const int rw = mt * 16 + r + hiq * 8;
          const float v = acc[r];
          if (M == 0)      Qs[wv][rw * 16 + lh] = (bf16)(v * 0.25f);
          else if (M == 1) Ks[wv][rw * 16 + lh] = (bf16)v;
          else             Vt[wv][lh * 64 + rw] = (bf16)v;
        }
      }
    }
    // ---- dots + bias + softmax ----
#pragma unroll
    for (int mt = 0; mt < 4; ++mt) {
      float e[4][8];
      const int rowq = mt * 16 + lh;
      const v16bf a = cat16(*(const bf16x8*)&Qs[wv][rowq * 16 + hiq * 8], bzero8());
#pragma unroll
      for (int nt = 0; nt < 4; ++nt) {
        v16bf bm;
        bm = cat16(*(const bf16x8*)&Ks[wv][(nt * 16 + lh) * 16 + hiq * 8], bzero8());
        v8f acc = wmma_bf(a, bm, fzero8());
#pragma unroll
        for (int r = 0; r < 8; ++r) {
          const int i = mt * 16 + r + hiq * 8;
          const int j = nt * 16 + lh;
          const int idx = ((i >> 3) - (j >> 3) + 7) * 15 + ((i & 7) - (j & 7) + 7);
          e[nt][r] = acc[r] + rpb[idx * 16 + h];
        }
      }
#pragma unroll
      for (int r = 0; r < 8; ++r) {
        float m = red16_max(fmaxf(fmaxf(e[0][r], e[1][r]), fmaxf(e[2][r], e[3][r])));
        float s = 0.f;
#pragma unroll
        for (int nt = 0; nt < 4; ++nt) { float t = __expf(e[nt][r] - m); e[nt][r] = t; s += t; }
        s = red16_sum(s);
        const float inv = 1.0f / s;
        const int i = mt * 16 + r + hiq * 8;
#pragma unroll
        for (int nt = 0; nt < 4; ++nt) At[wv][i * 64 + nt * 16 + lh] = (bf16)(e[nt][r] * inv);
      }
    }
    // ---- out = attn @ V ----
#pragma unroll
    for (int mt = 0; mt < 4; ++mt) {
      v8f acc = fzero8();
      const int i0 = mt * 16 + lh;
#pragma unroll
      for (int kc = 0; kc < 2; ++kc) {
        const int base = kc * 32 + hiq * 8;
        v16bf a = ld_a16(&At[wv][i0 * 64 + base], &At[wv][i0 * 64 + base + 16]);
        v16bf bm = ld_a16(&Vt[wv][lh * 64 + base], &Vt[wv][lh * 64 + base + 16]);
        acc = wmma_bf(a, bm, acc);
      }
      const int c = h * 16 + lh;
#pragma unroll
      for (int r = 0; r < 8; ++r) {
        const int i = mt * 16 + r + hiq * 8;
        attn[(((size_t)b * 256 + c) * 256 + (h0 + (i >> 3))) * 256 + (w0 + (i & 7))] = (bf16)acc[r];
      }
    }
  }
}

// ================= K2: local = bn(conv3x3) + bn(conv1x1), implicit GEMM =================
__global__ void __launch_bounds__(256) fa_local(const float* __restrict__ y,
                                                const bf16* __restrict__ pl1,
                                                const bf16* __restrict__ pl2,
                                                const float* __restrict__ cloc,
                                                bf16* __restrict__ localb) {
  __shared__ __attribute__((aligned(32))) bf16 Ys[3 * 66 * 64];
  const int tid = threadIdx.x;
  const int wv = tid >> 5, lane = tid & 31, lh = lane & 15, hiq = (lane >> 4) & 1;
  const int g = blockIdx.x;
  const int wseg = g & 3, h = (g >> 2) & 255, b = g >> 10;
  const int nt0 = wv * 2, nt1 = wv * 2 + 1;

  v8f acc[2][4];
#pragma unroll
  for (int j = 0; j < 2; ++j)
#pragma unroll
    for (int mt = 0; mt < 4; ++mt) acc[j][mt] = fzero8();

  for (int c0 = 0; c0 < 256; c0 += 64) {
    __syncthreads();
    // interior columns: vectorized float4 (16B aligned)
    for (int t = tid; t < 3072; t += 256) {
      const int f4i = t & 15, row = t >> 4;
      const int ci = row & 63, dy = row >> 6;
      const int gy = h + dy - 1;
      const int px0 = wseg * 64 + f4i * 4;
      float4 v = make_float4(0.f, 0.f, 0.f, 0.f);
      if (gy >= 0 && gy < 256)
        v = *(const float4*)(y + (((size_t)b * 256 + c0 + ci) * 256 + gy) * 256 + px0);
      bf16* d = &Ys[(dy * 66 + 1 + f4i * 4) * 64 + ci];
      d[0 * 64] = (bf16)v.x; d[1 * 64] = (bf16)v.y;
      d[2 * 64] = (bf16)v.z; d[3 * 64] = (bf16)v.w;
    }
    // edge columns 0 and 65
    for (int t = tid; t < 384; t += 256) {
      const int side = t & 1, row = t >> 1;
      const int ci = row & 63, dy = row >> 6;
      const int gy = h + dy - 1;
      const int col = side ? 65 : 0;
      const int px = wseg * 64 + col - 1;
      float v = 0.f;
      if (gy >= 0 && gy < 256 && px >= 0 && px < 256)
        v = y[(((size_t)b * 256 + c0 + ci) * 256 + gy) * 256 + px];
      Ys[(dy * 66 + col) * 64 + ci] = (bf16)v;
    }
    // prefetch next channel chunk
    if (c0 + 64 < 256 && tid < 64)
      __builtin_prefetch(y + (((size_t)b * 256 + c0 + 64 + tid) * 256 + h) * 256 + wseg * 64, 0, 0);
    __syncthreads();
#pragma unroll
    for (int tap = 0; tap < 9; ++tap) {
      const int tdy = tap / 3, tdx = tap - tdy * 3;
#pragma unroll
      for (int kloc = 0; kloc < 2; ++kloc) {
        const int kt = (c0 >> 5) + kloc;
        v16bf b0 = *(const v16bf*)(pl1 + ((size_t)((tap * 16 + nt0) * 8 + kt) * 32 + lane) * 16);
        v16bf b1 = *(const v16bf*)(pl1 + ((size_t)((tap * 16 + nt1) * 8 + kt) * 32 + lane) * 16);
        const int base = kloc * 32 + hiq * 8;
#pragma unroll
        for (int mt = 0; mt < 4; ++mt) {
          const int col = mt * 16 + lh + tdx;
          v16bf a = ld_a16(&Ys[(tdy * 66 + col) * 64 + base], &Ys[(tdy * 66 + col) * 64 + base + 16]);
          acc[0][mt] = wmma_bf(a, b0, acc[0][mt]);
          acc[1][mt] = wmma_bf(a, b1, acc[1][mt]);
        }
      }
    }
#pragma unroll
    for (int kloc = 0; kloc < 2; ++kloc) {
      const int kt = (c0 >> 5) + kloc;
      v16bf b0 = *(const v16bf*)(pl2 + ((size_t)(nt0 * 8 + kt) * 32 + lane) * 16);
      v16bf b1 = *(const v16bf*)(pl2 + ((size_t)(nt1 * 8 + kt) * 32 + lane) * 16);
      const int base = kloc * 32 + hiq * 8;
#pragma unroll
      for (int mt = 0; mt < 4; ++mt) {
        const int col = mt * 16 + lh + 1;
        v16bf a = ld_a16(&Ys[(66 + col) * 64 + base], &Ys[(66 + col) * 64 + base + 16]);
        acc[0][mt] = wmma_bf(a, b0, acc[0][mt]);
        acc[1][mt] = wmma_bf(a, b1, acc[1][mt]);
      }
    }
  }
#pragma unroll
  for (int j = 0; j < 2; ++j) {
    const int ch = (wv * 2 + j) * 16 + lh;
    const float cadd = cloc[ch];
#pragma unroll
    for (int mt = 0; mt < 4; ++mt)
#pragma unroll
      for (int r = 0; r < 8; ++r) {
        const int p = mt * 16 + r + hiq * 8;
        localb[(((size_t)b * 256 + ch) * 256 + h) * 256 + wseg * 64 + p] = (bf16)(acc[j][mt][r] + cadd);
      }
  }
}

// ================= K3: out = avgpool_v + avgpool_h + local =================
__global__ void __launch_bounds__(256) fa_pool(const bf16* __restrict__ attn,
                                               const bf16* __restrict__ localb,
                                               bf16* __restrict__ sum) {
  const int id = blockIdx.x * 256 + threadIdx.x;
  const int w = id & 255, h = (id >> 8) & 255, c = (id >> 16) & 255, b = id >> 24;
  const size_t base = (((size_t)b * 256 + c) * 256) * 256;
  float sx = 0.f, sy = 0.f;
#pragma unroll
  for (int j = 0; j < 8; ++j) {
    const int rp = h - 3 + j;
    if (rp >= 0 && rp <= 256) sx += (float)attn[base + (size_t)(rp == 256 ? 254 : rp) * 256 + w];
    const int cp = w - 3 + j;
    if (cp >= 0 && cp <= 256) sy += (float)attn[base + (size_t)h * 256 + (cp == 256 ? 254 : cp)];
  }
  const float v = 0.125f * (sx + sy) + (float)localb[base + (size_t)h * 256 + w];
  sum[base + (size_t)h * 256 + w] = (bf16)v;
}

// ================= K4: depthwise 8x8 + BN -> GEMM-ready [pixel][channel] layout =================
__global__ void __launch_bounds__(256) fa_dw(const bf16* __restrict__ sum,
                                             const float* __restrict__ pdw,
                                             const float* __restrict__ cdw,
                                             bf16* __restrict__ dwt) {
  __shared__ bf16 Hs[16 * 8 * 72];
  __shared__ float Wl[16 * 64];
  const int g = blockIdx.x;
  const int wseg = g & 3, h = (g >> 2) & 255, cg = (g >> 10) & 15, b = g >> 14;

  for (int t = threadIdx.x; t < 1024; t += 256)
    Wl[t] = pdw[(cg * 16 + (t >> 6)) * 64 + (t & 63)];

  for (int t = threadIdx.x; t < 16 * 8 * 71; t += 256) {
    const int cx = t % 71, r2 = t / 71;
    const int ky = r2 & 7, ci = r2 >> 3;
    const int gy = h + ky - 3, gx = wseg * 64 + cx - 3;
    float v = 0.f;
    if (gy >= 0 && gy <= 256 && gx >= 0 && gx <= 256) {
      const int my = (gy == 256) ? 254 : gy;
      const int mx = (gx == 256) ? 254 : gx;
      v = (float)sum[(((size_t)b * 256 + cg * 16 + ci) * 256 + my) * 256 + mx];
    }
    Hs[(ci * 8 + ky) * 72 + cx] = (bf16)v;
  }
  __syncthreads();

#pragma unroll
  for (int k = 0; k < 4; ++k) {
    const int t = threadIdx.x + k * 256;
    const int ci = t & 15, p = t >> 4;
    float a = 0.f;
#pragma unroll
    for (int ky = 0; ky < 8; ++ky)
#pragma unroll
      for (int kx = 0; kx < 8; ++kx)
        a += Wl[ci * 64 + ky * 8 + kx] * (float)Hs[(ci * 8 + ky) * 72 + p + kx];
    const int c = cg * 16 + ci;
    dwt[(((size_t)(b * 256 + h) * 4 + wseg) * 64 + p) * 256 + c] = (bf16)(a + cdw[c]);
  }
}

// ================= K5: pointwise 1x1 (256x256 GEMM) -> fp32 output =================
__global__ void __launch_bounds__(256) fa_pw(const bf16* __restrict__ dwt,
                                             const bf16* __restrict__ ppw,
                                             float* __restrict__ out) {
  __shared__ __attribute__((aligned(32))) bf16 Ds[64 * 256];
  const int tid = threadIdx.x;
  const int wv = tid >> 5, lane = tid & 31, lh = lane & 15, hiq = (lane >> 4) & 1;
  const int g = blockIdx.x;
  const int wseg = g & 3, h = (g >> 2) & 255, b = g >> 10;

  if (tid == 0) __builtin_prefetch(ppw, 0, 0);

  // linear 32KB stage: dwt block is already in [p][ci] order
  const bf16* gsrc = dwt + ((size_t)(b * 256 + h) * 4 + wseg) * 16384;
#ifdef FA_ASYNC
  {
    as1_v4i_p gp = (as1_v4i_p)gsrc;
    as3_v4i_p lp = (as3_v4i_p)&Ds[0];
    for (int t = tid; t < 2048; t += 256)
      __builtin_amdgcn_global_load_async_to_lds_b128(gp + t, lp + t, 0, 0);
    __builtin_amdgcn_s_wait_asynccnt(0);
  }
#else
  for (int t = tid; t < 2048; t += 256)
    *(int4*)((char*)&Ds[0] + t * 16) = *(const int4*)((const char*)gsrc + (size_t)t * 16);
#endif
  __syncthreads();

  const int nt0 = wv * 2, nt1 = wv * 2 + 1;
  v8f acc[2][4];
#pragma unroll
  for (int j = 0; j < 2; ++j)
#pragma unroll
    for (int mt = 0; mt < 4; ++mt) acc[j][mt] = fzero8();
#pragma unroll
  for (int kt = 0; kt < 8; ++kt) {
    v16bf b0 = *(const v16bf*)(ppw + ((size_t)(nt0 * 8 + kt) * 32 + lane) * 16);
    v16bf b1 = *(const v16bf*)(ppw + ((size_t)(nt1 * 8 + kt) * 32 + lane) * 16);
    const int base = kt * 32 + hiq * 8;
#pragma unroll
    for (int mt = 0; mt < 4; ++mt) {
      const int row = mt * 16 + lh;
      v16bf a = ld_a16(&Ds[row * 256 + base], &Ds[row * 256 + base + 16]);
      acc[0][mt] = wmma_bf(a, b0, acc[0][mt]);
      acc[1][mt] = wmma_bf(a, b1, acc[1][mt]);
    }
  }
#pragma unroll
  for (int j = 0; j < 2; ++j) {
    const int ch = (wv * 2 + j) * 16 + lh;
#pragma unroll
    for (int mt = 0; mt < 4; ++mt)
#pragma unroll
      for (int r = 0; r < 8; ++r) {
        const int p = mt * 16 + r + hiq * 8;
        out[(((size_t)b * 256 + ch) * 256 + h) * 256 + wseg * 64 + p] = acc[j][mt][r];
      }
  }
}

extern "C" void kernel_launch(void* const* d_in, const int* in_sizes, int n_in,
                              void* d_out, int out_size, void* d_ws, size_t ws_size,
                              hipStream_t stream) {
  const float* x   = (const float*)d_in[0];
  const float* y   = (const float*)d_in[1];
  const float* wqkv= (const float*)d_in[2];
  const float* wl1 = (const float*)d_in[3];
  const float* g1  = (const float*)d_in[4];
  const float* b1  = (const float*)d_in[5];
  const float* m1  = (const float*)d_in[6];
  const float* v1  = (const float*)d_in[7];
  const float* wl2 = (const float*)d_in[8];
  const float* g2  = (const float*)d_in[9];
  const float* b2  = (const float*)d_in[10];
  const float* m2  = (const float*)d_in[11];
  const float* v2  = (const float*)d_in[12];
  const float* wdw = (const float*)d_in[13];
  const float* gp  = (const float*)d_in[14];
  const float* bp  = (const float*)d_in[15];
  const float* mp  = (const float*)d_in[16];
  const float* vp  = (const float*)d_in[17];
  const float* wpw = (const float*)d_in[18];
  const float* rpb = (const float*)d_in[19];

  char* ws = (char*)d_ws;
  bf16* attn   = (bf16*)(ws + OFF_ATTN);
  bf16* localb = (bf16*)(ws + OFF_LOCAL);
  bf16* sum    = (bf16*)(ws + OFF_SUM);
  bf16* dwt    = (bf16*)(ws + OFF_DW);
  bf16* pqkv   = (bf16*)(ws + OFF_WQKV);
  bf16* pl1    = (bf16*)(ws + OFF_WL1);
  bf16* pl2    = (bf16*)(ws + OFF_WL2);
  bf16* ppw    = (bf16*)(ws + OFF_WPW);
  float* pdw   = (float*)(ws + OFF_WDW);
  float* cloc  = (float*)(ws + OFF_CLOC);
  float* cdw   = (float*)(ws + OFF_CDW);

  const int packN = 768 * 256 + 256 * 256 * 9 + 256 * 256 + 256 * 256 + 256 * 64 + 256;
  fa_pack<<<(packN + 255) / 256, 256, 0, stream>>>(wqkv, wl1, g1, b1, m1, v1, wl2, g2, b2, m2, v2,
                                                   wdw, gp, bp, mp, vp, wpw,
                                                   pqkv, pl1, pl2, ppw, pdw, cloc, cdw);
  fa_attn<<<BB * 32 * 32, 64, 0, stream>>>(x, pqkv, rpb, attn);
  fa_local<<<BB * HH * 4, 256, 0, stream>>>(y, pl1, pl2, cloc, localb);
  fa_pool<<<(int)(NELM / 256), 256, 0, stream>>>(attn, localb, sum);
  fa_dw<<<BB * DIM * 16 * 16 * 2, 256, 0, stream>>>(sum, pdw, cdw, dwt);
  fa_pw<<<BB * HH * 4, 256, 0, stream>>>(dwt, ppw, (float*)d_out);
}
